// FlatTensorRouter_8186207666953
// MI455X (gfx1250) — matrix operand order repaired and checked by reading, not compiled, not run on hardware
//
#include <hip/hip_runtime.h>

// ---------------------------------------------------------------------------
// MoE router for MI455X (gfx1250, wave32).
// GEMM (16384x4096 @ 4096x64) via split-bf16 WMMA:
//   x*w ~= x_hi*W_hi + x_hi*W_lo + x_lo*W_hi   (3x v_wmma_f32_16x16x32_bf16)
// keeps ~fp32 accuracy while using the bf16 matrix pipe, so the kernel stays
// HBM-bound (~11us for 256MB of x at 23.3 TB/s) instead of being throttled by
// the slow f32 WMMA path. W is pre-split into bf16 hi/lo (1MB, L2-resident).
// Each wave computes 16 tokens x all 64 experts -> x read from HBM once.
// ---------------------------------------------------------------------------

#define D_MODEL        4096
#define N_EXPERTS      64
#define N_TOKENS       16384
#define WAVES_PER_BLK  4
#define TOK_PER_WAVE   16
#define TOK_PER_BLK    (WAVES_PER_BLK * TOK_PER_WAVE)   // 64
#define N_BLOCKS       (N_TOKENS / TOK_PER_BLK)         // 256

typedef __bf16 bf16_t;
typedef __attribute__((ext_vector_type(16))) __bf16 v16bf;
typedef __attribute__((ext_vector_type(8)))  float  v8f;

// ---- Prolog: split W (f32) into bf16 hi + lo residual -----------------------
__global__ __launch_bounds__(256) void moe_prep_kernel(
    const float* __restrict__ W, bf16_t* __restrict__ Whi, bf16_t* __restrict__ Wlo) {
  int i = blockIdx.x * 256 + threadIdx.x;           // covers 64*4096 exactly
  float w = W[i];
  bf16_t h = (bf16_t)w;                             // RNE f32->bf16
  Whi[i] = h;
  Wlo[i] = (bf16_t)(w - (float)h);                  // residual
}

// ---- Main: logits GEMM + top-2 softmax + per-block expert prob sums --------
__global__ __launch_bounds__(128) void moe_router_kernel(
    const float*  __restrict__ x,
    const bf16_t* __restrict__ Whi,
    const bf16_t* __restrict__ Wlo,
    float* __restrict__ out_w,      // [N_TOKENS][2]
    float* __restrict__ out_i,      // [N_TOKENS][2] (indices stored as f32)
    float* __restrict__ partials)   // [N_BLOCKS][N_EXPERTS]
{
  __shared__ float lds_l[WAVES_PER_BLK][TOK_PER_WAVE][N_EXPERTS]; // logits->probs
  __shared__ float lds_rden[WAVES_PER_BLK][TOK_PER_WAVE];
  __shared__ float lds_ws[WAVES_PER_BLK][N_EXPERTS];

  const int tid  = threadIdx.x;
  const int wave = tid >> 5;
  const int lane = tid & 31;
  const int half = lane >> 4;        // 0: lanes 0-15, 1: lanes 16-31
  const int l16  = lane & 15;

  const int tok0 = (blockIdx.x * WAVES_PER_BLK + wave) * TOK_PER_WAVE;

  // ---------------- Phase 1: K-loop GEMM -----------------------------------
  // A 16x32 bf16 (this code's K order): lane<16 row=lane holds K 0..15;
  // lane>=16 holds K 16..31. Element j -> K = base + j (16 contiguous floats).
  // B 32x16 bf16: lane = expert column; lane<16 K=0..15, lane>=16 K=16..31,
  // element j -> K = base + j  (contiguous in W's d dimension).
  v8f acc[4] = {v8f{}, v8f{}, v8f{}, v8f{}};
  const float* xr = x + (size_t)(tok0 + l16) * D_MODEL;
  const int ac = half ? 16 : 0;      // A per-lane K chunk base
  const int bc = half ? 16 : 0;      // B per-lane K chunk base

  for (int d0 = 0; d0 < D_MODEL; d0 += 32) {
    const float4* p0 = (const float4*)(xr + d0 + ac);
    const float4* p1 = (const float4*)(xr + d0 + ac + 8);
    float4 f0 = p0[0], f1 = p0[1], f2 = p1[0], f3 = p1[1];
    float fv[16] = {f0.x,f0.y,f0.z,f0.w, f1.x,f1.y,f1.z,f1.w,
                    f2.x,f2.y,f2.z,f2.w, f3.x,f3.y,f3.z,f3.w};
    v16bf a_hi, a_lo;
#pragma unroll
    for (int j = 0; j < 16; ++j) {
      bf16_t h = (bf16_t)fv[j];
      a_hi[j] = h;
      a_lo[j] = (bf16_t)(fv[j] - (float)h);
    }
#pragma unroll
    for (int t = 0; t < 4; ++t) {
      const size_t boff = (size_t)(t * 16 + l16) * D_MODEL + d0 + bc;
      const v16bf bh = *(const v16bf*)(Whi + boff);   // 32B aligned, L2-hot
      const v16bf bl = *(const v16bf*)(Wlo + boff);
      v8f a = acc[t];
      a = __builtin_amdgcn_wmma_f32_16x16x32_bf16(false, a_hi, false, bh, (short)0, a, false, false);
      a = __builtin_amdgcn_wmma_f32_16x16x32_bf16(false, a_hi, false, bl, (short)0, a, false, false);
      a = __builtin_amdgcn_wmma_f32_16x16x32_bf16(false, a_lo, false, bh, (short)0, a, false, false);
      acc[t] = a;
    }
  }

  // C/D layout: VGPR v, lanes0-15 -> M=v, lanes16-31 -> M=v+8; N = lane&15.
#pragma unroll
  for (int t = 0; t < 4; ++t)
#pragma unroll
    for (int v = 0; v < 8; ++v)
      lds_l[wave][half * 8 + v][t * 16 + l16] = acc[t][v];

  __syncthreads();

  // ---------------- Phase 2: per-token top-2 + softmax ---------------------
  if (lane < 16) {
    const int m = lane;
    float* lrow = lds_l[wave][m];
    float m1 = -3.0e38f, m2 = -3.0e38f; int i1 = 0, i2 = 0;
    for (int e = 0; e < N_EXPERTS; ++e) {         // strict '>' => lowest index on ties
      float v = lrow[e];
      if (v > m1)      { m2 = m1; i2 = i1; m1 = v; i1 = e; }
      else if (v > m2) { m2 = v;  i2 = e; }
    }
    float den = 0.f;
    for (int e = 0; e < N_EXPERTS; ++e) {
      float p = expf(lrow[e] - m1);
      lrow[e] = p;                                 // overwrite logits with exp()
      den += p;
    }
    lds_rden[wave][m] = 1.0f / den;

    float e2 = expf(m2 - m1);
    float w1 = 1.0f / (1.0f + e2);                 // softmax over {m1, m2}
    const int gt = tok0 + m;
    out_w[gt * 2 + 0] = w1;
    out_w[gt * 2 + 1] = e2 * w1;
    out_i[gt * 2 + 0] = (float)i1;
    out_i[gt * 2 + 1] = (float)i2;
  }
  __syncthreads();

  // ---------------- Phase 3: deterministic expert prob sums ----------------
#pragma unroll
  for (int q = 0; q < 2; ++q) {                    // 32 lanes x 2 experts = 64
    const int e = lane * 2 + q;
    float s = 0.f;
    for (int m = 0; m < TOK_PER_WAVE; ++m)
      s += lds_l[wave][m][e] * lds_rden[wave][m];
    lds_ws[wave][e] = s;
  }
  __syncthreads();
  if (tid < N_EXPERTS) {
    float s = lds_ws[0][tid] + lds_ws[1][tid] + lds_ws[2][tid] + lds_ws[3][tid];
    partials[(size_t)blockIdx.x * N_EXPERTS + tid] = s;   // no atomics
  }
}

// ---- Epilog: aux_loss = N_EXPERTS * sum(mean_probs^2), fixed-order reduce --
__global__ __launch_bounds__(64) void moe_aux_kernel(
    const float* __restrict__ partials, float* __restrict__ aux_out) {
  __shared__ float red[N_EXPERTS];
  const int e = threadIdx.x;
  float s = 0.f;
  for (int b = 0; b < N_BLOCKS; ++b) s += partials[(size_t)b * N_EXPERTS + e];
  float mean = s * (1.0f / (float)N_TOKENS);
  red[e] = mean * mean;
  __syncthreads();
  for (int off = 32; off > 0; off >>= 1) {
    if (e < off) red[e] += red[e + off];
    __syncthreads();
  }
  if (e == 0) aux_out[0] = red[0] * (float)N_EXPERTS;
}

extern "C" void kernel_launch(void* const* d_in, const int* in_sizes, int n_in,
                              void* d_out, int out_size, void* d_ws, size_t ws_size,
                              hipStream_t stream) {
  const float* x = (const float*)d_in[0];   // (4,4096,4096) f32
  const float* W = (const float*)d_in[1];   // (64,4096) f32

  float* out   = (float*)d_out;
  float* out_w = out;                        // 32768 top-k weights
  float* out_i = out + N_TOKENS * 2;         // 32768 top-k indices (as f32)
  float* aux   = out + N_TOKENS * 4;         // 1 aux loss

  char* ws = (char*)d_ws;
  const size_t wElems = (size_t)N_EXPERTS * D_MODEL;
  bf16_t* Whi      = (bf16_t*)ws;                                   // 512 KB
  bf16_t* Wlo      = (bf16_t*)(ws + wElems * sizeof(bf16_t));       // 512 KB
  float*  partials = (float*)(ws + 2 * wElems * sizeof(bf16_t));    // 64 KB

  moe_prep_kernel<<<(N_EXPERTS * D_MODEL) / 256, 256, 0, stream>>>(W, Whi, Wlo);
  moe_router_kernel<<<N_BLOCKS, WAVES_PER_BLK * 32, 0, stream>>>(
      x, Whi, Wlo, out_w, out_i, partials);
  moe_aux_kernel<<<1, N_EXPERTS, 0, stream>>>(partials, aux);
}
